// SgmAttention_56066503082061
// MI455X (gfx1250) — matrix operand order repaired and checked by reading, not compiled, not run on hardware
//
#include <hip/hip_runtime.h>
#include <hip/hip_bf16.h>

typedef __attribute__((ext_vector_type(16))) __bf16 v16bf;
typedef __attribute__((ext_vector_type(8)))  float  v8f;
typedef unsigned int   u32;
typedef unsigned short u16;
typedef __attribute__((ext_vector_type(4)))  u32    u32x4;
typedef __attribute__((ext_vector_type(2)))  u32    u32x2;

#define ENC_H 512
#define DEC_H 512
#define ATT_H 512
#define BATCH 64
#define SEQ   2048
#define PITCH 520   // bf16 elements per LDS row (512 + 8 pad -> fragment rows hit distinct banks)

__device__ __forceinline__ u32 bf16_rne(float f) {
    u32 x = __float_as_uint(f);
    return (x + 0x7FFFu + ((x >> 16) & 1u)) >> 16;
}
__device__ __forceinline__ u32 pack_bf16x2(float lo, float hi) {
    return bf16_rne(lo) | (bf16_rne(hi) << 16);
}

union FragAB { v16bf v; u32x4 q[2]; };

// ---------------------------------------------------------------------------
// Kernel 0a: convert U_w [512,512] f32 -> bf16 row-major (reduction dim contiguous,
// which is exactly the B-fragment layout: N = row a, K = col e).
// ---------------------------------------------------------------------------
__global__ void __launch_bounds__(256)
cvt_u_kernel(const float* __restrict__ U, u32* __restrict__ Ubf2) {
    int i = blockIdx.x * 256 + threadIdx.x;          // over 131072 float pairs
    float2 f = ((const float2*)U)[i];
    Ubf2[i] = pack_bf16x2(f.x, f.y);
}

// ---------------------------------------------------------------------------
// Kernel 0b: state_term[b][a] = dot(s[b,:], W_w[a,:]) + W_b[a]   (64x512, K=512)
// ---------------------------------------------------------------------------
__global__ void __launch_bounds__(256)
state_term_kernel(const float* __restrict__ s, const float* __restrict__ W_w,
                  const float* __restrict__ W_b, float* __restrict__ state) {
    extern __shared__ char dyn_smem[];
    float* s_lds = (float*)dyn_smem;                 // [64][513]
    const int t = threadIdx.x;
    for (int i = t; i < BATCH * DEC_H; i += 256) {
        int bb = i >> 9, e = i & 511;
        s_lds[bb * 513 + e] = s[i];
    }
    __syncthreads();
    const int a    = blockIdx.x * 32 + (t >> 3);
    const int bset = t & 7;
    float acc[8] = {0,0,0,0,0,0,0,0};
    const float* wrow = W_w + (size_t)a * DEC_H;
    for (int e = 0; e < DEC_H; ++e) {
        float w = wrow[e];
        #pragma unroll
        for (int j = 0; j < 8; ++j)
            acc[j] += w * s_lds[(bset + 8 * j) * 513 + e];
    }
    float wb = W_b[a];
    #pragma unroll
    for (int j = 0; j < 8; ++j)
        state[(bset + 8 * j) * ATT_H + a] = acc[j] + wb;
}

// ---------------------------------------------------------------------------
// U-tile staging helpers (4 x b128 per thread = one 16x512 bf16 tile per block)
// ---------------------------------------------------------------------------
__device__ __forceinline__ void load_u_tile(const u16* __restrict__ Ubf, int a0,
                                            int tid, u32x4 st[4]) {
    #pragma unroll
    for (int j = 0; j < 4; ++j) {
        int i  = tid + j * 256;
        int r  = i >> 6;
        int c8 = (i & 63) * 8;
        st[j] = *(const u32x4*)(Ubf + (size_t)(a0 + r) * ENC_H + c8);
    }
}
__device__ __forceinline__ void store_u_tile(u16* __restrict__ dst, int tid,
                                             const u32x4 st[4]) {
    #pragma unroll
    for (int j = 0; j < 4; ++j) {
        int i  = tid + j * 256;
        int r  = i >> 6;
        int c8 = (i & 63) * 8;
        *(u32x4*)(dst + r * PITCH + c8) = st[j];
    }
}

// ---------------------------------------------------------------------------
// Kernel 1: fused  logits = V . tanh(context @ U^T + state + U_b)
// 1024 blocks x 8 waves; wave owns a 16-row context tile (bf16 in LDS);
// U column-tiles are double-buffered: next tile's global loads issue before the
// WMMA K-loop (latency hidden under 16 x v_wmma), stored to the alternate LDS
// buffer after compute -> one barrier per iteration, no load stall.
// ---------------------------------------------------------------------------
__global__ void __launch_bounds__(256)
attn_logits_wmma(const float* __restrict__ ctx,      // [B*S, 512] f32
                 const u16*   __restrict__ Ubf,      // [512, 512] bf16
                 const float* __restrict__ state,    // [B, 512]
                 const float* __restrict__ U_b,      // [512]
                 const float* __restrict__ V_w,      // [512]
                 const float* __restrict__ V_b,      // [1]
                 float* __restrict__ logits)         // [B*S]
{
    extern __shared__ char dyn_smem[];
    u16* bufA    = (u16*)dyn_smem;                   // U tile buffer 0: 16 x PITCH
    u16* bufB    = bufA + 16 * PITCH;                // U tile buffer 1
    u16* ctx_all = bufB + 16 * PITCH;                // 8 waves x 16 x PITCH

    const int tid  = threadIdx.x;
    const int wave = tid >> 5;
    const int lane = tid & 31;
    const int h    = lane >> 4;                      // half-wave select (ISA layout)
    const int ln   = lane & 15;

    u16* my_ctx = ctx_all + wave * (16 * PITCH);
    const int row0 = blockIdx.x * 128 + wave * 16;   // 16 rows for this wave
    const int b    = (blockIdx.x * 128) >> 11;       // whole block inside one batch

    // stage & convert this wave's 16x512 context tile to bf16 (row-major, padded)
    {
        const float* src = ctx + (size_t)row0 * ENC_H;
        for (int i = lane; i < 16 * 128; i += 32) {
            int m  = i >> 7;
            int ep = (i & 127) * 4;
            float4 f = *(const float4*)(src + (size_t)m * ENC_H + ep);
            u32x2 pk;
            pk.x = pack_bf16x2(f.x, f.y);
            pk.y = pack_bf16x2(f.z, f.w);
            *(u32x2*)(my_ctx + m * PITCH + ep) = pk;
        }
    }

    // prologue: stage U tile 0 into buffer A
    u32x4 stage[4];
    load_u_tile(Ubf, 0, tid, stage);
    store_u_tile(bufA, tid, stage);
    __syncthreads();

    float partial[8] = {0,0,0,0,0,0,0,0};
    const float* stateb = state + b * ATT_H;

    for (int at = 0; at < 32; ++at) {
        const int a0 = at * 16;
        const u16* cur = (at & 1) ? bufB : bufA;
        u16*       nxt = (at & 1) ? bufA : bufB;

        // issue next tile's global loads now; they retire under the WMMAs below
        if (at + 1 < 32) load_u_tile(Ubf, a0 + 16, tid, stage);
        if (at + 2 < 32) __builtin_prefetch(Ubf + (size_t)(a0 + 32) * ENC_H, 0, 1);

        v8f acc = {0,0,0,0,0,0,0,0};
        // A frag (16-bit A 16x32): lane ln(+16h): M=ln, K = k0+8h+{0..7}, k0+8h+16+{0..7}
        const u16* arow = my_ctx + ln * PITCH + h * 8;
        // B frag (16-bit B 32x16): lane ln(+16h): N=ln, K = k0+8h+{0..7}, k0+8h+16+{0..7}
        const u16* brow = cur + ln * PITCH + h * 8;
        #pragma unroll
        for (int k0 = 0; k0 < ENC_H; k0 += 32) {
            FragAB A, Bf;
            A.q[0]  = *(const u32x4*)(arow + k0);
            A.q[1]  = *(const u32x4*)(arow + k0 + 16);
            Bf.q[0] = *(const u32x4*)(brow + k0);
            Bf.q[1] = *(const u32x4*)(brow + k0 + 16);
            acc = __builtin_amdgcn_wmma_f32_16x16x32_bf16(
                      false, A.v, false, Bf.v, (short)0, acc, false, false);
        }

        // epilogue for this column tile: lane's column a = a0+ln for all 8 D rows
        const int ag   = a0 + ln;
        const float sA = stateb[ag] + U_b[ag];
        const float vw = V_w[ag];
        #pragma unroll
        for (int r = 0; r < 8; ++r)
            partial[r] += vw * tanhf(acc[r] + sA);   // D vgpr r: M = r + 8h, N = ln

        // commit next U tile to the alternate buffer; single barrier per iter
        if (at + 1 < 32) store_u_tile(nxt, tid, stage);
        __syncthreads();
    }

    // reduce the 16 columns held across each 16-lane half; xor masks 1,2,4,8 stay in-half
    const float vb = V_b[0];
    #pragma unroll
    for (int r = 0; r < 8; ++r) {
        float p = partial[r];
        p += __shfl_xor(p, 1);
        p += __shfl_xor(p, 2);
        p += __shfl_xor(p, 4);
        p += __shfl_xor(p, 8);
        if (ln == 0)
            logits[row0 + h * 8 + r] = p + vb;       // lane0 -> rows 0..7, lane16 -> rows 8..15
    }
}

// ---------------------------------------------------------------------------
// Kernel 2: row softmax, in place. 64 blocks x 256 threads, 8 elems/thread.
// ---------------------------------------------------------------------------
__global__ void __launch_bounds__(256)
softmax_rows(float* __restrict__ logits) {
    __shared__ float red[8];
    const int t = threadIdx.x, wave = t >> 5, lane = t & 31;
    float* row = logits + blockIdx.x * SEQ;
    float v[8]; float m = -3.4e38f;
    #pragma unroll
    for (int i = 0; i < 8; ++i) { v[i] = row[t + i * 256]; m = fmaxf(m, v[i]); }
    #pragma unroll
    for (int off = 1; off < 32; off <<= 1) m = fmaxf(m, __shfl_xor(m, off));
    if (lane == 0) red[wave] = m;
    __syncthreads();
    float bm = red[0];
    #pragma unroll
    for (int i = 1; i < 8; ++i) bm = fmaxf(bm, red[i]);
    float s = 0.f;
    #pragma unroll
    for (int i = 0; i < 8; ++i) { v[i] = __expf(v[i] - bm); s += v[i]; }
    #pragma unroll
    for (int off = 1; off < 32; off <<= 1) s += __shfl_xor(s, off);
    __syncthreads();
    if (lane == 0) red[wave] = s;
    __syncthreads();
    float bs = 0.f;
    #pragma unroll
    for (int i = 0; i < 8; ++i) bs += red[i];
    float inv = 1.f / bs;
    #pragma unroll
    for (int i = 0; i < 8; ++i) row[t + i * 256] = v[i] * inv;
}

// ---------------------------------------------------------------------------
// Kernel 3a: partial c_t over S-chunks of 256 -> 512 blocks streaming 256 MB.
// part[b][ch][e] = sum_{s in chunk} w[b,s] * context[b,s,e]
// ---------------------------------------------------------------------------
__global__ void __launch_bounds__(256)
ct_partial_kernel(const float* __restrict__ ctx, const float* __restrict__ w,
                  float* __restrict__ part) {     // [64][8][512]
    __shared__ float wrow[256];
    const int b  = blockIdx.x >> 3;
    const int ch = blockIdx.x & 7;
    const int t  = threadIdx.x;
    wrow[t] = w[b * SEQ + ch * 256 + t];
    __syncthreads();
    const float* base = ctx + ((size_t)b * SEQ + ch * 256) * ENC_H + t * 2;
    float a0 = 0.f, a1 = 0.f;
    #pragma unroll 4
    for (int si = 0; si < 256; ++si) {
        float2 c = *(const float2*)(base + (size_t)si * ENC_H);
        float ww = wrow[si];
        a0 += ww * c.x;
        a1 += ww * c.y;
    }
    float* dst = part + ((size_t)b * 8 + ch) * ENC_H + t * 2;
    dst[0] = a0;
    dst[1] = a1;
}

// Kernel 3b: deterministic reduction of the 8 S-chunk partials.
__global__ void __launch_bounds__(256)
ct_reduce_kernel(const float* __restrict__ part, float* __restrict__ out) {
    int i = blockIdx.x * 256 + threadIdx.x;        // over 64*512
    int b = i >> 9, e = i & 511;
    const float* p = part + ((size_t)b * 8) * ENC_H + e;
    float s = 0.f;
    #pragma unroll
    for (int c = 0; c < 8; ++c) s += p[c * ENC_H];
    out[i] = s;
}

// ---------------------------------------------------------------------------
extern "C" void kernel_launch(void* const* d_in, const int* in_sizes, int n_in,
                              void* d_out, int out_size, void* d_ws, size_t ws_size,
                              hipStream_t stream) {
    (void)in_sizes; (void)n_in; (void)out_size; (void)ws_size;
    const float* s   = (const float*)d_in[0];
    const float* ctx = (const float*)d_in[1];
    const float* U_w = (const float*)d_in[2];
    const float* U_b = (const float*)d_in[3];
    const float* W_w = (const float*)d_in[4];
    const float* W_b = (const float*)d_in[5];
    const float* V_w = (const float*)d_in[6];
    const float* V_b = (const float*)d_in[7];
    float* out = (float*)d_out;

    char* ws = (char*)d_ws;
    u16*   Ubf    = (u16*)ws;                                   // 524288 B
    float* state  = (float*)(ws + 524288);                      // 131072 B
    float* logits = (float*)(ws + 524288 + 131072);             // 524288 B
    float* part   = (float*)(ws + 524288 + 131072 + 524288);    // 1048576 B

    cvt_u_kernel<<<512, 256, 0, stream>>>(U_w, (u32*)Ubf);
    state_term_kernel<<<16, 256, 64 * 513 * 4, stream>>>(s, W_w, W_b, state);
    const size_t lds_main = (size_t)(16 * PITCH * 2) * 10;      // 2 U buffers + 8 ctx tiles
    attn_logits_wmma<<<1024, 256, lds_main, stream>>>(ctx, Ubf, state, U_b, V_w, V_b, logits);
    softmax_rows<<<64, 256, 0, stream>>>(logits);
    ct_partial_kernel<<<512, 256, 0, stream>>>(ctx, logits, part);
    ct_reduce_kernel<<<128, 256, 0, stream>>>(part, out);
}
